// BiLSTM_79963701117082
// MI455X (gfx1250) — matrix operand order repaired and checked by reading, not compiled, not run on hardware
//
#include <hip/hip_runtime.h>

#define Tseq 16384
#define HID  128

typedef __attribute__((ext_vector_type(16))) __bf16 v16bf;
typedef __attribute__((ext_vector_type(8)))  float  v8f;

__device__ inline unsigned short f2bf_bits(float f) {
  union { float f; unsigned u; } v; v.f = f;
  unsigned r = v.u + 0x7FFFu + ((v.u >> 16) & 1u);   // round-to-nearest-even
  return (unsigned short)(r >> 16);
}
__device__ inline __bf16 bits2bf(unsigned short s) {
  return __builtin_bit_cast(__bf16, s);
}
__device__ inline __bf16 f2bf(float f) { return bits2bf(f2bf_bits(f)); }
__device__ inline float sigm(float x) { return 1.f / (1.f + __expf(-x)); }

__device__ inline void sched_fence() {
#if __has_builtin(__builtin_amdgcn_sched_barrier)
  __builtin_amdgcn_sched_barrier(0);
#endif
}

// ---------------------------------------------------------------------------
// Recurrence kernel: one workgroup per direction (grid.x = 2), 32 waves.
// Gate-row remap: wave w owns W_hh rows {128*g + 4w + j : g in 0..3, j in 0..3},
// i.e. all four gates for h-indices 4w..4w+3. After the 4-WMMA chain the whole
// i/f/g/o -> c/h update is wave-local (in-wave LDS round-trip, no barrier).
// h is double-buffered in LDS -> exactly ONE workgroup barrier per step.
// ---------------------------------------------------------------------------
__global__ __launch_bounds__(1024) void lstm_rec_kernel(
    const float* __restrict__ x,                                   // [T] (layer0) or null
    const float* __restrict__ gxF, const float* __restrict__ gxR,  // [T*512] (layer1) or null
    const float* __restrict__ wihF, const float* __restrict__ wihR,// [512] (layer0 col) or null
    const float* __restrict__ whhF, const float* __restrict__ whhR,// [512*128]
    const float* __restrict__ bihF, const float* __restrict__ bhhF,
    const float* __restrict__ bihR, const float* __restrict__ bhhR,
    float* __restrict__ hout)                                      // [T*256], cols dir*128..+128
{
  const int dir = blockIdx.x;                    // 0 = forward, 1 = reverse
  const float* gx  = dir ? gxR  : gxF;
  const float* wih = dir ? wihR : wihF;
  const float* whh = dir ? whhR : whhF;
  const float* bih = dir ? bihR : bihF;
  const float* bhh = dir ? bhhR : bhhF;

  __shared__ unsigned short h_bf[2][HID];        // double-buffered h (bf16 bits)
  __shared__ float gates_s[4 * HID];             // wave-private 16-float slices
  __shared__ float wcol_s[4 * HID];              // layer0 input weight column
  __shared__ float bsum_s[4 * HID];              // b_ih + b_hh (layer0 path)

  const int tid  = threadIdx.x;
  const int wave = tid >> 5;
  const int lane = tid & 31;
  const int half = lane >> 4;
  const int m    = lane & 15;

  if (tid < 4 * HID) {
    bsum_s[tid] = bih[tid] + bhh[tid];
    wcol_s[tid] = wih ? wih[tid] : 0.f;
  }
  if (tid < HID) h_bf[0][tid] = 0;

  // Persistent A-fragments. Gate-interleaved row map:
  //   local row m -> global W_hh row 128*(m>>2) + 4*wave + (m&3)
  // 16-bit A 16x32 layout: element e -> K = (e>>3)*16 + half*8 + (e&7).
  const int arow = 128 * (m >> 2) + 4 * wave + (m & 3);
  v16bf a_frag[4];
#pragma unroll
  for (int kt = 0; kt < 4; ++kt) {
#pragma unroll
    for (int e = 0; e < 16; ++e) {
      const int K = kt * 32 + (e >> 3) * 16 + half * 8 + (e & 7);
      a_frag[kt][e] = f2bf(whh[arow * HID + K]);
    }
  }

  const int gbase = wave * 16;                   // wave-private gates slice
  const int j     = 4 * wave + lane;             // h index handled by lanes 0..3
  float creg = 0.f;                              // cell state (lanes 0..3 of each wave)
  __syncthreads();

  for (int s = 0; s < Tseq; ++s) {
    const int t    = dir ? (Tseq - 1 - s) : s;
    const int rbuf = s & 1, wbuf = rbuf ^ 1;

    // Prefetch next step's gx row into cache (overlaps the WMMA chain).
    if (gx && (s + 1 < Tseq) && tid < 4 * HID) {
      const int tn = dir ? (t - 1) : (t + 1);
      __builtin_prefetch(&gx[(size_t)tn * 4 * HID + tid], 0, 1);
    }

    // B-fragments: 16-bit B 32x16 layout, element e -> K = (e>>3)*16 + half*8 + (e&7).
    // h replicated across all N columns -> every column of D equals W_hh@h.
    v16bf b_frag[4];
#pragma unroll
    for (int kt = 0; kt < 4; ++kt) {
#pragma unroll
      for (int e = 0; e < 16; ++e)
        b_frag[kt][e] = bits2bf(h_bf[rbuf][kt * 32 + (e >> 3) * 16 + half * 8 + (e & 7)]);
    }
    sched_fence();                               // keep all ds_loads issued first

    v8f acc = {0.f, 0.f, 0.f, 0.f, 0.f, 0.f, 0.f, 0.f};
#pragma unroll
    for (int kt = 0; kt < 4; ++kt)
      acc = __builtin_amdgcn_wmma_f32_16x16x32_bf16(
          false, a_frag[kt], false, b_frag[kt], (short)0, acc, false, false);

    // Publish this wave's 16 dot products (wave-private slice; columns are
    // identical, lanes 0 and 16 carry M=0..7 / M=8..15 in acc[0..7]).
    if (lane == 0 || lane == 16) {
#pragma unroll
      for (int r = 0; r < 8; ++r) gates_s[gbase + (half << 3) + r] = acc[r];
    }

    // Wave-local update: same-wave LDS ops are in-order -> no barrier needed.
    if (lane < 4) {
      float gi = gates_s[gbase + lane];
      float gf = gates_s[gbase + 4 + lane];
      float gg = gates_s[gbase + 8 + lane];
      float go = gates_s[gbase + 12 + lane];
      if (gx) {
        const float* g = &gx[(size_t)t * 4 * HID];
        gi += g[j]; gf += g[HID + j]; gg += g[2 * HID + j]; go += g[3 * HID + j];
      } else {
        const float xi = x[t];
        gi += xi * wcol_s[j]           + bsum_s[j];
        gf += xi * wcol_s[HID + j]     + bsum_s[HID + j];
        gg += xi * wcol_s[2 * HID + j] + bsum_s[2 * HID + j];
        go += xi * wcol_s[3 * HID + j] + bsum_s[3 * HID + j];
      }
      const float iv = sigm(gi), fv = sigm(gf), gv = tanhf(gg), ov = sigm(go);
      creg = fv * creg + iv * gv;
      const float h = ov * tanhf(creg);
      h_bf[wbuf][j] = f2bf_bits(h);
      hout[(size_t)t * (2 * HID) + dir * HID + j] = h;
    }
    __syncthreads();                             // sole barrier: h[wbuf] visible
  }
}

// ---------------------------------------------------------------------------
// Throughput GEMM: C[M][N] = A[M][Ka] * W[N][Ka]^T + bias (+ optional leaky).
// One 16x16 output tile per wave, K in 32-chunks of v_wmma_f32_16x16x32_bf16.
// ---------------------------------------------------------------------------
__global__ __launch_bounds__(256) void gemm_wmma_bf16(
    const float* __restrict__ A, const float* __restrict__ W,
    const float* __restrict__ bias0, const float* __restrict__ bias1,
    float* __restrict__ C, int M, int N, int Ka, int leaky)
{
  const int lane = threadIdx.x & 31;
  const int half = lane >> 4;
  const int l15  = lane & 15;
  const int ntiles = N >> 4;
  const int tiles  = (M >> 4) * ntiles;
  const int wg = blockIdx.x * (blockDim.x >> 5) + (threadIdx.x >> 5);
  if (wg >= tiles) return;                        // wave-uniform exit
  const int tm = wg / ntiles, tn = wg % ntiles;
  const int arow = tm * 16 + l15;
  const int bcol = tn * 16 + l15;

  v8f acc = {0.f, 0.f, 0.f, 0.f, 0.f, 0.f, 0.f, 0.f};
  for (int kb = 0; kb < Ka; kb += 32) {
    v16bf a_frag, b_frag;
#pragma unroll
    for (int e = 0; e < 16; ++e) {
      const int ka = kb + (e >> 3) * 16 + half * 8 + (e & 7);
      a_frag[e] = f2bf(A[(size_t)arow * Ka + ka]);
      const int kk = kb + (e >> 3) * 16 + half * 8 + (e & 7);
      b_frag[e] = f2bf(W[(size_t)bcol * Ka + kk]);
    }
    acc = __builtin_amdgcn_wmma_f32_16x16x32_bf16(
        false, a_frag, false, b_frag, (short)0, acc, false, false);
  }

  const float bv = bias0[bcol] + (bias1 ? bias1[bcol] : 0.f);
#pragma unroll
  for (int r = 0; r < 8; ++r) {
    const int row = tm * 16 + half * 8 + r;
    float v = acc[r] + bv;
    if (leaky) v = v > 0.f ? v : 0.01f * v;
    C[(size_t)row * N + bcol] = v;
  }
}

// ---------------------------------------------------------------------------
// FC2: y[t] = dot(fc2_w, h2[t]) + b2. One wave32 per timestep.
// ---------------------------------------------------------------------------
__global__ __launch_bounds__(256) void fc2_kernel(
    const float* __restrict__ h2, const float* __restrict__ w2,
    const float* __restrict__ b2, float* __restrict__ y)
{
  const int wv   = blockIdx.x * (blockDim.x >> 5) + (threadIdx.x >> 5);
  const int lane = threadIdx.x & 31;
  if (wv >= Tseq) return;
  float s = 0.f;
#pragma unroll
  for (int k = 0; k < 4; ++k)
    s += h2[(size_t)wv * HID + lane + 32 * k] * w2[lane + 32 * k];
#pragma unroll
  for (int off = 16; off > 0; off >>= 1) s += __shfl_down(s, off, 32);
  if (lane == 0) y[wv] = s + b2[0];
}

extern "C" void kernel_launch(void* const* d_in, const int* in_sizes, int n_in,
                              void* d_out, int out_size, void* d_ws, size_t ws_size,
                              hipStream_t stream) {
  const float* x     = (const float*)d_in[0];
  const float* wih0  = (const float*)d_in[1];
  const float* whh0  = (const float*)d_in[2];
  const float* bih0  = (const float*)d_in[3];
  const float* bhh0  = (const float*)d_in[4];
  const float* wih0r = (const float*)d_in[5];
  const float* whh0r = (const float*)d_in[6];
  const float* bih0r = (const float*)d_in[7];
  const float* bhh0r = (const float*)d_in[8];
  const float* wih1  = (const float*)d_in[9];
  const float* whh1  = (const float*)d_in[10];
  const float* bih1  = (const float*)d_in[11];
  const float* bhh1  = (const float*)d_in[12];
  const float* wih1r = (const float*)d_in[13];
  const float* whh1r = (const float*)d_in[14];
  const float* bih1r = (const float*)d_in[15];
  const float* bhh1r = (const float*)d_in[16];
  const float* fc1w  = (const float*)d_in[17];
  const float* fc1b  = (const float*)d_in[18];
  const float* fc2w  = (const float*)d_in[19];
  const float* fc2b  = (const float*)d_in[20];

  // Workspace layout (floats): all intermediates fit in MI455X's 192MB L2.
  float* h0   = (float*)d_ws;                     // [T][256]
  float* h1   = h0   + (size_t)Tseq * 256;        // [T][256]
  float* gx1f = h1   + (size_t)Tseq * 256;        // [T][512]
  float* gx1r = gx1f + (size_t)Tseq * 512;        // [T][512]
  float* h2   = gx1r + (size_t)Tseq * 512;        // [T][128]

  // Layer 0 recurrence (IN=1: input projection folded into the loop).
  lstm_rec_kernel<<<2, 1024, 0, stream>>>(x, nullptr, nullptr, wih0, wih0r,
      whh0, whh0r, bih0, bhh0, bih0r, bhh0r, h0);

  // Layer 1 input projections: gx1 = h0 @ W_ih^T + (b_ih + b_hh), both dirs.
  {
    const int tiles  = (Tseq / 16) * (512 / 16);
    const int blocks = (tiles + 7) / 8;
    gemm_wmma_bf16<<<blocks, 256, 0, stream>>>(h0, wih1,  bih1,  bhh1,  gx1f, Tseq, 512, 256, 0);
    gemm_wmma_bf16<<<blocks, 256, 0, stream>>>(h0, wih1r, bih1r, bhh1r, gx1r, Tseq, 512, 256, 0);
  }

  // Layer 1 recurrence (gx already carries the biases).
  lstm_rec_kernel<<<2, 1024, 0, stream>>>(nullptr, gx1f, gx1r, nullptr, nullptr,
      whh1, whh1r, bih1, bhh1, bih1r, bhh1r, h1);

  // FC1 + leaky ReLU.
  {
    const int tiles  = (Tseq / 16) * (128 / 16);
    const int blocks = (tiles + 7) / 8;
    gemm_wmma_bf16<<<blocks, 256, 0, stream>>>(h1, fc1w, fc1b, nullptr, h2, Tseq, 128, 256, 1);
  }

  // FC2 head.
  fc2_kernel<<<(Tseq * 32 + 255) / 256, 256, 0, stream>>>(h2, fc2w, fc2b, (float*)d_out);
}
